// _EncoderBlock_17978733101381
// MI455X (gfx1250) — matrix operand order repaired and checked by reading, not compiled, not run on hardware
//
#include <hip/hip_runtime.h>
#include <math.h>

typedef __bf16 bf16;
typedef __attribute__((ext_vector_type(16))) __bf16 v16bf;
typedef __attribute__((ext_vector_type(8)))  __bf16 v8bf;
typedef __attribute__((ext_vector_type(8)))  float  v8f;

#define EPS 1e-6f
constexpr int kB  = 8;
constexpr int kL  = 2048;
constexpr int kD  = 512;
constexpr int kM  = 256;     // feature dim M
constexpr int kD2 = 1024;    // 2*D
constexpr int kH  = 2048;    // 4*D
constexpr int kR  = kB * kL; // 16384 rows
constexpr float kScl      = 0.1767766952966369f; // (2D)^-0.25
constexpr float kInvSqrtM = 0.0625f;             // 1/sqrt(256)

#define BM 128
#define BN 128
#define BK 32

// Async global->LDS copy of 32 contiguous bytes per lane (2x B128).
// INST_OFFSET applies to both the LDS and global address (ISA 08 §4.4).
__device__ __forceinline__ void async_copy_32B(unsigned lds_off, const bf16* g) {
    unsigned long long ga = (unsigned long long)g;
    asm volatile("global_load_async_to_lds_b128 %0, %1, off\n\t"
                 "global_load_async_to_lds_b128 %0, %1, off offset:16"
                 :: "v"(lds_off), "v"(ga) : "memory");
}
__device__ __forceinline__ void wait_async() {
    asm volatile("s_wait_asynccnt 0" ::: "memory");
}

// ---------------------------------------------------------------------------
// Tiled bf16 WMMA GEMM:
//   C[M,N] (or C^T[N,M] if TRANS_OUT) = epilogue( alpha * A[M,K] @ Bt[N,K]^T )
//   A row-major (M,K); Bt is B TRANSPOSED, row-major (N,K)  -> both tiles
//   stage into LDS with contiguous 16B async copies, no in-LDS transpose.
//   Double-buffered LDS, one barrier per K-step, async staging overlapped
//   with v_wmma_f32_16x16x32_bf16 on the other buffer.
//   Epilogue: v = alpha*dot - row_sub[row]; v += bias[col]; relu; v /= (row_div[row]+EPS)
// ---------------------------------------------------------------------------
template<bool OUT_BF16, bool RELU, bool TRANS_OUT>
__global__ __launch_bounds__(256)
void gemm_bf16_wmma(const bf16* __restrict__ A, const bf16* __restrict__ Bt,
                    void* __restrict__ Cout,
                    int M, int N, int K, int lda, int ldb, int ldc,
                    long long sA, long long sB, long long sC,
                    float alpha,
                    const float* __restrict__ row_sub,
                    const float* __restrict__ row_div,
                    int rowvec_stride,
                    const float* __restrict__ bias)
{
    __shared__ alignas(16) bf16 As[2][BM][BK];   // 2 x 8KB
    __shared__ alignas(16) bf16 Bs[2][BN][BK];   // 2 x 8KB (n-major: Bs[n][k])

    const int bz = blockIdx.z;
    A  += (long long)bz * sA;
    Bt += (long long)bz * sB;

    const int tid   = threadIdx.x;
    const int wid   = tid >> 5;
    const int lane  = tid & 31;
    const int wm    = wid & 3;    // 4 waves * 32 rows
    const int wn    = wid >> 2;   // 2 waves * 64 cols
    const int tileM = blockIdx.y * BM;
    const int tileN = blockIdx.x * BN;
    const int lrow  = lane & 15;  // m (A) / n (B,C) within 16
    const int lhalf = lane >> 4;  // K-half-group select

    v8f zero = {0.f,0.f,0.f,0.f,0.f,0.f,0.f,0.f};
    v8f acc[2][4];
    #pragma unroll
    for (int i = 0; i < 2; ++i)
        #pragma unroll
        for (int j = 0; j < 4; ++j) acc[i][j] = zero;

    // staging: each thread owns one 16-element (32B) chunk of each tile
    const int srow = tid >> 1;          // 0..127
    const int scol = (tid & 1) * 16;    // 0 / 16
    const bf16* gA = A  + (long long)(tileM + srow) * lda + scol;
    const bf16* gB = Bt + (long long)(tileN + srow) * ldb + scol;

    auto stage = [&](int buf, int k0) {
        // low 32 bits of a generic shared pointer are the LDS byte offset
        unsigned la = (unsigned)(unsigned long long)&As[buf][srow][scol];
        unsigned lb = (unsigned)(unsigned long long)&Bs[buf][srow][scol];
        async_copy_32B(la, gA + k0);
        async_copy_32B(lb, gB + k0);
    };

    auto compute = [&](int buf) {
        // B fragments: lane n = lrow, halves hold K = 16*lhalf + 0..15
        union BF { v16bf v; v8bf h[2]; } bfrag[4];
        #pragma unroll
        for (int in = 0; in < 4; ++in) {
            const int bc = wn * 64 + in * 16 + lrow;
            bfrag[in].h[0] = *(const v8bf*)&Bs[buf][bc][16 * lhalf];
            bfrag[in].h[1] = *(const v8bf*)&Bs[buf][bc][16 * lhalf + 8];
        }
        #pragma unroll
        for (int im = 0; im < 2; ++im) {
            // A fragment: lane m = lrow, halves hold K = 16*lhalf + 0..15
            const int ar = wm * 32 + im * 16 + lrow;
            union AF { v16bf v; v8bf h[2]; } afrag;
            afrag.h[0] = *(const v8bf*)&As[buf][ar][16 * lhalf];
            afrag.h[1] = *(const v8bf*)&As[buf][ar][16 * lhalf + 8];
            #pragma unroll
            for (int in = 0; in < 4; ++in) {
                acc[im][in] = __builtin_amdgcn_wmma_f32_16x16x32_bf16(
                    false, afrag.v, false, bfrag[in].v,
                    (short)0, acc[im][in], false, false);
            }
        }
    };

    int buf = 0;
    stage(0, 0);
    wait_async();
    __syncthreads();
    for (int k0 = 0; k0 < K; k0 += BK) {
        const int nxt = k0 + BK;
        if (nxt < K) {
            stage(buf ^ 1, nxt);
            if (nxt + BK < K) {   // far prefetch -> global_prefetch_b8
                __builtin_prefetch(gA + nxt + BK, 0, 1);
                __builtin_prefetch(gB + nxt + BK, 0, 1);
            }
        }
        compute(buf);
        if (nxt < K) {
            wait_async();       // our async writes into buf^1 have landed
            __syncthreads();    // everyone done reading buf / writing buf^1
        }
        buf ^= 1;
    }

    // epilogue: C layout — lane n = lane&15, vgpr j -> m = 8*(lane>>4)+j
    float* cf = (float*)Cout;
    bf16*  cb = (bf16*)Cout;
    const long long cbase = (long long)bz * sC;
    const int rvb = bz * rowvec_stride;
    #pragma unroll
    for (int im = 0; im < 2; ++im) {
        #pragma unroll
        for (int j = 0; j < 8; ++j) {
            const int row = tileM + wm * 32 + im * 16 + lhalf * 8 + j;
            const float rs = row_sub ? row_sub[rvb + row] : 0.f;
            const float rd = row_div ? 1.f / (row_div[rvb + row] + EPS) : 1.f;
            #pragma unroll
            for (int in = 0; in < 4; ++in) {
                const int col = tileN + wn * 64 + in * 16 + lrow;
                float v = acc[im][in][j] * alpha - rs;
                if (bias) v += bias[col];
                if (RELU) v = fmaxf(v, 0.f);
                v *= rd;
                const long long idx = TRANS_OUT
                    ? (cbase + (long long)col * ldc + row)
                    : (cbase + (long long)row * ldc + col);
                if (OUT_BF16) cb[idx] = (bf16)v;
                else          cf[idx] = v;
            }
        }
    }
}

// ---------------------------------------------------------------------------
// Packing / elementwise kernels
// ---------------------------------------------------------------------------
__global__ void k_cvt_bf16(const float* __restrict__ in, bf16* __restrict__ out, int n) {
    int i = blockIdx.x * blockDim.x + threadIdx.x;
    if (i < n) out[i] = (bf16)in[i];
}

// Wt[n,k] = W[k,n]  (f32 (Kdim,Ndim) row-major -> bf16 (Ndim,Kdim) row-major)
__global__ void k_pack_wT(const float* __restrict__ in, bf16* __restrict__ out,
                          int Kdim, int Ndim) {
    int i = blockIdx.x * blockDim.x + threadIdx.x;
    if (i >= Kdim * Ndim) return;
    int n = i / Kdim, k = i % Kdim;
    out[i] = (bf16)in[k * Ndim + n];
}

// x = [zr | zi] as bf16, (R, 2D)
__global__ void k_pack_x(const float* __restrict__ zr, const float* __restrict__ zi,
                         bf16* __restrict__ out, int total) {
    int i = blockIdx.x * blockDim.x + threadIdx.x;
    if (i >= total) return;
    int r = i / kD2, c = i % kD2;
    float v = (c < kD) ? zr[r * kD + c] : zi[r * kD + (c - kD)];
    out[i] = (bf16)v;
}

// Wc1^T (2H x 2D): Wc1 = [[W1r, W1i], [-W1i, W1r]] (2D x 2H)
__global__ void k_pack_wc1T(const float* __restrict__ w1r, const float* __restrict__ w1i,
                            bf16* __restrict__ out, int total) {
    int i = blockIdx.x * blockDim.x + threadIdx.x;
    if (i >= total) return;
    int n = i / kD2, k = i % kD2;   // n: 0..2H-1 (cols of Wc1), k: 0..2D-1 (rows)
    float v;
    if (k < kD) v = (n < kH) ? w1r[k * kH + n] : w1i[k * kH + (n - kH)];
    else {
        int k2 = k - kD;
        v = (n < kH) ? -w1i[k2 * kH + n] : w1r[k2 * kH + (n - kH)];
    }
    out[i] = (bf16)v;
}

// Wc2^T (2D x 2H): Wc2 = [[W2r, W2i], [-W2i, W2r]] (2H x 2D)
__global__ void k_pack_wc2T(const float* __restrict__ w2r, const float* __restrict__ w2i,
                            bf16* __restrict__ out, int total) {
    int i = blockIdx.x * blockDim.x + threadIdx.x;
    if (i >= total) return;
    int n = i / (2 * kH), k = i % (2 * kH);  // n: cols of Wc2 (0..2D-1), k: rows (0..2H-1)
    float v;
    if (k < kH) v = (n < kD) ? w2r[k * kD + n] : w2i[k * kD + (n - kD)];
    else {
        int k2 = k - kH;
        v = (n < kD) ? -w2i[k2 * kD + n] : w2r[k2 * kD + (n - kD)];
    }
    out[i] = (bf16)v;
}

__global__ void k_cat_f32(const float* __restrict__ a, const float* __restrict__ b,
                          float* __restrict__ out, int na, int nb) {
    int i = blockIdx.x * blockDim.x + threadIdx.x;
    if (i < na + nb) out[i] = (i < na) ? a[i] : b[i - na];
}

// out[r] = 0.5 * sum_c A[r,c]^2   (block per row)
__global__ void k_rowsumsq(const bf16* __restrict__ A, float* __restrict__ out, int cols) {
    __shared__ float red[256];
    const int r = blockIdx.x;
    const bf16* p = A + (long long)r * cols;
    float acc = 0.f;
    for (int c = threadIdx.x; c < cols; c += 256) {
        float v = (float)p[c];
        acc += v * v;
    }
    red[threadIdx.x] = acc;
    __syncthreads();
    for (int s = 128; s > 0; s >>= 1) {
        if (threadIdx.x < s) red[threadIdx.x] += red[threadIdx.x + s];
        __syncthreads();
    }
    if (threadIdx.x == 0) out[r] = 0.5f * red[0];
}

// phi = exp(p - rowmax(p)) * scale, bf16 out (block per row)
__global__ void k_phi_rowmax(const float* __restrict__ pq, bf16* __restrict__ phi,
                             int cols, float scale) {
    __shared__ float red[256];
    const int r = blockIdx.x;
    const float* p = pq + (long long)r * cols;
    float m = -3.4e38f;
    for (int c = threadIdx.x; c < cols; c += 256) m = fmaxf(m, p[c]);
    red[threadIdx.x] = m;
    __syncthreads();
    for (int s = 128; s > 0; s >>= 1) {
        if (threadIdx.x < s) red[threadIdx.x] = fmaxf(red[threadIdx.x], red[threadIdx.x + s]);
        __syncthreads();
    }
    const float mx = red[0];
    for (int c = threadIdx.x; c < cols; c += 256)
        phi[(long long)r * cols + c] = (bf16)(__expf(p[c] - mx) * scale);
}

__global__ void k_max_part(const float* __restrict__ in, float* __restrict__ part, long long n) {
    __shared__ float red[256];
    float m = -3.4e38f;
    for (long long i = (long long)blockIdx.x * 256 + threadIdx.x; i < n;
         i += (long long)gridDim.x * 256)
        m = fmaxf(m, in[i]);
    red[threadIdx.x] = m;
    __syncthreads();
    for (int s = 128; s > 0; s >>= 1) {
        if (threadIdx.x < s) red[threadIdx.x] = fmaxf(red[threadIdx.x], red[threadIdx.x + s]);
        __syncthreads();
    }
    if (threadIdx.x == 0) part[blockIdx.x] = red[0];
}

__global__ void k_max_final(const float* __restrict__ part, float* __restrict__ gmax, int n) {
    __shared__ float red[256];
    float m = -3.4e38f;
    for (int i = threadIdx.x; i < n; i += 256) m = fmaxf(m, part[i]);
    red[threadIdx.x] = m;
    __syncthreads();
    for (int s = 128; s > 0; s >>= 1) {
        if (threadIdx.x < s) red[threadIdx.x] = fmaxf(red[threadIdx.x], red[threadIdx.x + s]);
        __syncthreads();
    }
    if (threadIdx.x == 0) gmax[0] = red[0];
}

__global__ void k_phi_global(const float* __restrict__ pk, const float* __restrict__ gmax,
                             bf16* __restrict__ phi, long long n, float scale) {
    long long i = (long long)blockIdx.x * blockDim.x + threadIdx.x;
    if (i < n) phi[i] = (bf16)(__expf(pk[i] - gmax[0]) * scale);
}

// ksum[b,m] = sum_l phikT[(b*M+m), l]   (block per (b,m) row of phikT)
__global__ void k_ksum(const bf16* __restrict__ phikT, float* __restrict__ ksum) {
    __shared__ float red[256];
    const int row = blockIdx.x;                 // b*kM + m
    const bf16* p = phikT + (long long)row * kL;
    float acc = 0.f;
    for (int l = threadIdx.x; l < kL; l += 256) acc += (float)p[l];
    red[threadIdx.x] = acc;
    __syncthreads();
    for (int s = 128; s > 0; s >>= 1) {
        if (threadIdx.x < s) red[threadIdx.x] += red[threadIdx.x + s];
        __syncthreads();
    }
    if (threadIdx.x == 0) ksum[row] = red[0];
}

// den[r] = dot(phiq[r,:], ksum[b,:])
__global__ void k_den(const bf16* __restrict__ phiq, const float* __restrict__ ksum,
                      float* __restrict__ den) {
    int r = blockIdx.x * blockDim.x + threadIdx.x;
    if (r >= kR) return;
    const int b = r / kL;
    const bf16* p = phiq + (long long)r * kM;
    const float* ks = ksum + b * kM;
    float acc = 0.f;
    for (int m = 0; m < kM; ++m) acc += (float)p[m] * ks[m];
    den[r] = acc;
}

// modrelu(z + y) -> f32 pair + bf16 concat for next GEMM
__global__ void k_modrelu1(const float* __restrict__ zr, const float* __restrict__ zi,
                           const float* __restrict__ yo, const float* __restrict__ battn,
                           float* __restrict__ z1r, float* __restrict__ z1i,
                           bf16* __restrict__ zcat) {
    int i = blockIdx.x * blockDim.x + threadIdx.x;
    if (i >= kR * kD) return;
    int r = i / kD, d = i % kD;
    float xr = zr[i] + yo[(long long)r * kD2 + d];
    float xi = zi[i] + yo[(long long)r * kD2 + kD + d];
    float mag = sqrtf(xr * xr + xi * xi + EPS);
    float s = fmaxf(mag + battn[d], 0.f) / mag;
    float orr = xr * s, oii = xi * s;
    z1r[i] = orr;
    z1i[i] = oii;
    zcat[(long long)r * kD2 + d] = (bf16)orr;
    zcat[(long long)r * kD2 + kD + d] = (bf16)oii;
}

// final modrelu -> stacked (B,L,D,2) f32 output
__global__ void k_modrelu2(const float* __restrict__ z1r, const float* __restrict__ z1i,
                           const float* __restrict__ f, const float* __restrict__ bffn,
                           float* __restrict__ out) {
    int i = blockIdx.x * blockDim.x + threadIdx.x;
    if (i >= kR * kD) return;
    int r = i / kD, d = i % kD;
    float xr = z1r[i] + f[(long long)r * kD2 + d];
    float xi = z1i[i] + f[(long long)r * kD2 + kD + d];
    float mag = sqrtf(xr * xr + xi * xi + EPS);
    float s = fmaxf(mag + bffn[d], 0.f) / mag;
    out[(long long)i * 2]     = xr * s;
    out[(long long)i * 2 + 1] = xi * s;
}

static inline int cdiv(int a, int b) { return (a + b - 1) / b; }

// ---------------------------------------------------------------------------
extern "C" void kernel_launch(void* const* d_in, const int* in_sizes, int n_in,
                              void* d_out, int out_size, void* d_ws, size_t ws_size,
                              hipStream_t stream)
{
    (void)in_sizes; (void)n_in; (void)out_size; (void)ws_size;

    const float* zr   = (const float*)d_in[0];
    const float* zi   = (const float*)d_in[1];
    const float* Wq   = (const float*)d_in[2];
    const float* Wk   = (const float*)d_in[3];
    const float* Wv   = (const float*)d_in[4];
    const float* Wo   = (const float*)d_in[5];
    const float* Wf   = (const float*)d_in[6];
    const float* b_attn = (const float*)d_in[7];
    const float* W1r  = (const float*)d_in[8];
    const float* W1i  = (const float*)d_in[9];
    const float* b1r  = (const float*)d_in[10];
    const float* b1i  = (const float*)d_in[11];
    const float* W2r  = (const float*)d_in[12];
    const float* W2i  = (const float*)d_in[13];
    const float* b2r  = (const float*)d_in[14];
    const float* b2i  = (const float*)d_in[15];
    const float* b_ffn = (const float*)d_in[16];

    // ---- workspace layout (peak ~370 MB, explicit lifetime reuse) ----
    char* ws = (char*)d_ws;
    size_t off = 0;
    auto alloc = [&](size_t bytes) -> char* {
        char* p = ws + off;
        off = (off + bytes + 255) & ~(size_t)255;
        return p;
    };

    bf16* wqT   = (bf16*)alloc((size_t)kD2 * kD2 * 2);
    bf16* wkT   = (bf16*)alloc((size_t)kD2 * kD2 * 2);
    bf16* wvT   = (bf16*)alloc((size_t)kD2 * kD2 * 2);
    bf16* woT   = (bf16*)alloc((size_t)kD2 * kD2 * 2);
    bf16* wf_bf = (bf16*)alloc((size_t)kM * kD2 * 2);   // Bt for pq/pk is Wf itself
    bf16* wc1T  = (bf16*)alloc((size_t)2 * kH * kD2 * 2);
    bf16* wc2T  = (bf16*)alloc((size_t)kD2 * 2 * kH * 2);
    float* b1cat = (float*)alloc((size_t)2 * kH * 4);
    float* b2cat = (float*)alloc((size_t)kD2 * 4);

    bf16* x_bf = (bf16*)alloc((size_t)kR * kD2 * 2);      // reused as y_bf later
    char* big_base = ws + off;                            // 128MB region reused as h_bf
    bf16*  q_bf  = (bf16*)alloc((size_t)kR * kD2 * 2);
    bf16*  k_bf  = (bf16*)alloc((size_t)kR * kD2 * 2);
    bf16*  vT_bf = (bf16*)alloc((size_t)kR * kD2 * 2);    // (b, D2, L)
    float* pq    = (float*)alloc((size_t)kR * kM * 4);    // (R, M)
    float* pkT   = (float*)alloc((size_t)kR * kM * 4);    // (b, M, L)
    bf16*  h_bf  = (bf16*)big_base;                       // (R, 2H) bf16 = 128MB

    float* ssq   = (float*)alloc((size_t)kR * 4);
    float* ssk   = (float*)alloc((size_t)kR * 4);
    float* part  = (float*)alloc(1024 * 4);
    float* gmax  = (float*)alloc(256);
    bf16*  phiq  = (bf16*)alloc((size_t)kR * kM * 2);     // (R, M)
    bf16*  phikT = (bf16*)alloc((size_t)kR * kM * 2);     // (b, M, L)
    float* ksum  = (float*)alloc((size_t)kB * kM * 4);
    float* den   = (float*)alloc((size_t)kR * 4);
    bf16*  kvT   = (bf16*)alloc((size_t)kB * kM * kD2 * 2); // (b, D2, M)
    float* yo    = (float*)alloc((size_t)kR * kD2 * 4);   // reused as f later
    float* fbuf  = yo;
    float* z1rb  = (float*)alloc((size_t)kR * kD * 4);
    float* z1ib  = (float*)alloc((size_t)kR * kD * 4);
    bf16*  zcat  = (bf16*)alloc((size_t)kR * kD2 * 2);
    bf16*  y_bf  = x_bf;

    const int TB = 256;
    dim3 blk(256);

    // ---- pack weights (transposed for the GEMM's Bt layout) ----
    k_pack_wT<<<cdiv(kD2 * kD2, TB), TB, 0, stream>>>(Wq, wqT, kD2, kD2);
    k_pack_wT<<<cdiv(kD2 * kD2, TB), TB, 0, stream>>>(Wk, wkT, kD2, kD2);
    k_pack_wT<<<cdiv(kD2 * kD2, TB), TB, 0, stream>>>(Wv, wvT, kD2, kD2);
    k_pack_wT<<<cdiv(kD2 * kD2, TB), TB, 0, stream>>>(Wo, woT, kD2, kD2);
    k_cvt_bf16<<<cdiv(kM * kD2, TB), TB, 0, stream>>>(Wf, wf_bf, kM * kD2);
    k_pack_wc1T<<<cdiv(2 * kH * kD2, TB), TB, 0, stream>>>(W1r, W1i, wc1T, 2 * kH * kD2);
    k_pack_wc2T<<<cdiv(kD2 * 2 * kH, TB), TB, 0, stream>>>(W2r, W2i, wc2T, kD2 * 2 * kH);
    k_cat_f32<<<cdiv(2 * kH, TB), TB, 0, stream>>>(b1r, b1i, b1cat, kH, kH);
    k_cat_f32<<<cdiv(kD2, TB), TB, 0, stream>>>(b2r, b2i, b2cat, kD, kD);
    k_pack_x<<<cdiv(kR * kD2, TB), TB, 0, stream>>>(zr, zi, x_bf, kR * kD2);

    // ---- q, k = scl * x @ Wq/Wk ;  vT[b] = (x[b] @ Wv)^T ----
    dim3 g_qk(kD2 / BN, kR / BM, 1);
    gemm_bf16_wmma<true, false, false><<<g_qk, blk, 0, stream>>>(
        x_bf, wqT, q_bf, kR, kD2, kD2, kD2, kD2, kD2, 0, 0, 0,
        kScl, nullptr, nullptr, 0, nullptr);
    gemm_bf16_wmma<true, false, false><<<g_qk, blk, 0, stream>>>(
        x_bf, wkT, k_bf, kR, kD2, kD2, kD2, kD2, kD2, 0, 0, 0,
        kScl, nullptr, nullptr, 0, nullptr);
    dim3 g_v(kD2 / BN, kL / BM, kB);
    gemm_bf16_wmma<true, false, true><<<g_v, blk, 0, stream>>>(
        x_bf, wvT, vT_bf, kL, kD2, kD2, kD2, kD2, /*ldc=*/kL,
        (long long)kL * kD2, 0, (long long)kD2 * kL,
        1.0f, nullptr, nullptr, 0, nullptr);

    // ---- pq = q @ Wf^T - 0.5||q||^2 ;  pkT[b] = (k[b] @ Wf^T - 0.5||k||^2)^T ----
    k_rowsumsq<<<kR, 256, 0, stream>>>(q_bf, ssq, kD2);
    k_rowsumsq<<<kR, 256, 0, stream>>>(k_bf, ssk, kD2);
    dim3 g_pq(kM / BN, kR / BM, 1);
    gemm_bf16_wmma<false, false, false><<<g_pq, blk, 0, stream>>>(
        q_bf, wf_bf, pq, kR, kM, kD2, kD2, kD2, kM, 0, 0, 0,
        1.0f, ssq, nullptr, 0, nullptr);
    dim3 g_pk(kM / BN, kL / BM, kB);
    gemm_bf16_wmma<false, false, true><<<g_pk, blk, 0, stream>>>(
        k_bf, wf_bf, pkT, kL, kM, kD2, kD2, kD2, /*ldc=*/kL,
        (long long)kL * kD2, 0, (long long)kM * kL,
        1.0f, ssk, nullptr, kL, nullptr);

    // ---- feature maps ----
    k_phi_rowmax<<<kR, 256, 0, stream>>>(pq, phiq, kM, kInvSqrtM);
    k_max_part<<<1024, 256, 0, stream>>>(pkT, part, (long long)kR * kM);
    k_max_final<<<1, 256, 0, stream>>>(part, gmax, 1024);
    k_phi_global<<<cdiv(kR * kM, TB), TB, 0, stream>>>(pkT, gmax, phikT,
                                                       (long long)kR * kM, kInvSqrtM);
    k_ksum<<<kB * kM, 256, 0, stream>>>(phikT, ksum);

    // ---- kvT[b] = (phik[b]^T @ v[b])^T   (M=256, N=D2, K=L) ----
    dim3 g_kv(kD2 / BN, kM / BM, kB);
    gemm_bf16_wmma<true, false, true><<<g_kv, blk, 0, stream>>>(
        phikT, vT_bf, kvT, kM, kD2, kL, kL, kL, /*ldc=*/kM,
        (long long)kM * kL, (long long)kD2 * kL, (long long)kD2 * kM,
        1.0f, nullptr, nullptr, 0, nullptr);

    // ---- den, then y[b] = (phiq[b] @ kv[b]) / (den + eps) ----
    k_den<<<cdiv(kR, TB), TB, 0, stream>>>(phiq, ksum, den);
    dim3 g_y(kD2 / BN, kL / BM, kB);
    gemm_bf16_wmma<true, false, false><<<g_y, blk, 0, stream>>>(
        phiq, kvT, y_bf, kL, kD2, kM, kM, kM, kD2,
        (long long)kL * kM, (long long)kD2 * kM, (long long)kL * kD2,
        1.0f, nullptr, den, kL, nullptr);

    // ---- yo = y @ Wo (f32) ----
    dim3 g_o(kD2 / BN, kR / BM, 1);
    gemm_bf16_wmma<false, false, false><<<g_o, blk, 0, stream>>>(
        y_bf, woT, yo, kR, kD2, kD2, kD2, kD2, kD2, 0, 0, 0,
        1.0f, nullptr, nullptr, 0, nullptr);

    // ---- modrelu residual 1 ----
    k_modrelu1<<<cdiv(kR * kD, TB), TB, 0, stream>>>(zr, zi, yo, b_attn, z1rb, z1ib, zcat);

    // ---- h = relu([z1r|z1i] @ Wc1 + b1cat) bf16 ----
    dim3 g_h(2 * kH / BN, kR / BM, 1);
    gemm_bf16_wmma<true, true, false><<<g_h, blk, 0, stream>>>(
        zcat, wc1T, h_bf, kR, 2 * kH, kD2, kD2, kD2, 2 * kH, 0, 0, 0,
        1.0f, nullptr, nullptr, 0, b1cat);

    // ---- f = h @ Wc2 + b2cat (f32) ----
    dim3 g_f(kD2 / BN, kR / BM, 1);
    gemm_bf16_wmma<false, false, false><<<g_f, blk, 0, stream>>>(
        h_bf, wc2T, fbuf, kR, kD2, 2 * kH, 2 * kH, 2 * kH, kD2, 0, 0, 0,
        1.0f, nullptr, nullptr, 0, b2cat);

    // ---- modrelu residual 2 -> stacked (B,L,D,2) output ----
    k_modrelu2<<<cdiv(kR * kD, TB), TB, 0, stream>>>(z1rb, z1ib, fbuf, b_ffn,
                                                     (float*)d_out);
}
